// PartCodeTransformer_47184510713982
// MI455X (gfx1250) — matrix operand order repaired and checked by reading, not compiled, not run on hardware
//
#include <hip/hip_runtime.h>
#include <hip/hip_bf16.h>
#include <math.h>

// ---------------------------------------------------------------------------
// Types for CDNA5 WMMA (wave32).
// ---------------------------------------------------------------------------
typedef __bf16        bf16x16 __attribute__((ext_vector_type(16)));
typedef __bf16        bf16x8  __attribute__((ext_vector_type(8)));
typedef float         v8f     __attribute__((ext_vector_type(8)));
typedef unsigned int  u32x4   __attribute__((ext_vector_type(4)));
typedef int           i32x4   __attribute__((ext_vector_type(4)));
typedef int           i32x8   __attribute__((ext_vector_type(8)));

#if defined(__has_builtin)
#if __has_builtin(__builtin_amdgcn_tensor_load_to_lds)
#define USE_TDM 1
#endif
#endif

static __device__ inline v8f v8f_zero() {
  v8f z = {0.f, 0.f, 0.f, 0.f, 0.f, 0.f, 0.f, 0.f};
  return z;
}

static __device__ inline v8f wmma_bf16(bf16x16 a, bf16x16 b, v8f c) {
  return __builtin_amdgcn_wmma_f32_16x16x32_bf16(false, a, false, b,
                                                 (short)0, c, false, false);
}

static __device__ inline float gelu_exact(float x) {
  return 0.5f * x * (1.f + erff(x * 0.70710678118654752440f));
}

// ---------------------------------------------------------------------------
// Tensor Data Mover: 2-D tile (bf16) global -> LDS.  D# per ISA 08 §8.3/8.4.
// rows_rem/row_len give tensor dims from the tile start (OOB rows zero-fill,
// which doubles as M-masking for short A tiles).  Issue from one wave only.
// 6-arg builtin: (u32x4 g0, i32x8 g1, i32x4 g2, i32x4 g3, i32x8 extra, i32).
// ---------------------------------------------------------------------------
#ifdef USE_TDM
static __device__ inline void tdm_load_2d(const __bf16* gbase,
                                          unsigned lds_addr, int rows_rem,
                                          int row_len, int stride_elems,
                                          int tile_w, int tile_h) {
  const unsigned long long ga = (unsigned long long)(size_t)gbase;
  u32x4 g0;
  g0[0] = 1u;                                        // count=1, user D#
  g0[1] = lds_addr;                                  // LDS byte address
  g0[2] = (unsigned)(ga & 0xFFFFFFFFu);              // global_addr[31:0]
  g0[3] = (unsigned)((ga >> 32) & 0x1FFFFFFu) | (2u << 30);  // addr hi | type=2
  i32x8 g1;
  g1[0] = 1 << 16;                                   // data_size = 2 bytes
  g1[1] = (row_len & 0xFFFF) << 16;                  // tensor_dim0[15:0]
  g1[2] = ((row_len >> 16) & 0xFFFF) | ((rows_rem & 0xFFFF) << 16);
  g1[3] = ((rows_rem >> 16) & 0xFFFF) | ((tile_w & 0xFFFF) << 16);
  g1[4] = tile_h & 0xFFFF;                           // tile_dim1 (tile_dim2=0)
  g1[5] = stride_elems;                              // tensor_dim0_stride lo
  g1[6] = 0;
  g1[7] = 0;
  i32x4 z4 = {0, 0, 0, 0};
  i32x8 z8 = {0, 0, 0, 0, 0, 0, 0, 0};
  __builtin_amdgcn_tensor_load_to_lds(g0, g1, z4, z4, z8, 0);
}
#endif

static __device__ inline unsigned lds_addr_of(const void* p) {
  return (unsigned)(size_t)p;
}

// ---------------------------------------------------------------------------
// Fragment loaders (contiguous 16B LDS reads, per ISA 16-bit WMMA layouts).
// A 16x32: lane m=L&15, K half h=L>>4: K in {8h..8h+7} U {16+8h..16+8h+7}.
// B 32x16 from K-transposed store [n][k]: lane n=L&15, K same as A (8h.. U 16+8h..).
// C/D f32: vgpr r, lane L -> M = r + (L>>4)*8, N = L&15.
// ---------------------------------------------------------------------------
static __device__ inline bf16x16 frag_a_lds(const __bf16* base, int ld) {
  const int lane = threadIdx.x & 31;
  const __bf16* row = base + (lane & 15) * ld + ((lane >> 4) << 3);
  bf16x8 lo = *(const bf16x8*)(row);
  bf16x8 hi = *(const bf16x8*)(row + 16);
  return __builtin_shufflevector(lo, hi, 0, 1, 2, 3, 4, 5, 6, 7, 8, 9, 10, 11,
                                 12, 13, 14, 15);
}

static __device__ inline bf16x16 frag_bt_lds(const __bf16* base, int ld) {
  const int lane = threadIdx.x & 31;
  const __bf16* row = base + (lane & 15) * ld + ((lane >> 4) << 3);
  bf16x8 lo = *(const bf16x8*)(row);
  bf16x8 hi = *(const bf16x8*)(row + 16);
  return __builtin_shufflevector(lo, hi, 0, 1, 2, 3, 4, 5, 6, 7, 8, 9, 10, 11,
                                 12, 13, 14, 15);
}

// ---------------------------------------------------------------------------
// WMMA GEMM: C[M,N] = epi(A[M,K]bf16 @ W[K,N]bf16 + bias [+ resid]).
// 256 threads (8 waves), block tile 128x128, BK=64, double buffered.
// A tile via TDM (wave 0 issues, TENSORcnt sync); W tile staged transposed
// ([n][k], pad 72) so B-fragment reads are contiguous.
// ---------------------------------------------------------------------------
template <bool GELU_ACT, bool RESID, typename CT>
__global__ __launch_bounds__(256) void gemm_wmma(
    const __bf16* __restrict__ A, const __bf16* __restrict__ W,
    const float* __restrict__ bias, const float* __restrict__ resid,
    CT* __restrict__ C, int M, int K, int N) {
  __shared__ __bf16 As[2][128 * 64];
  __shared__ __bf16 Bs[2][128 * 72];
  const int tid  = threadIdx.x;
  const int lane = tid & 31;
  const int wave = tid >> 5;
  const int bm = blockIdx.y * 128;
  const int bn = blockIdx.x * 128;
  const int wm = (wave >> 1) * 32;
  const int wn = (wave & 1) * 64;

  v8f acc[2][4];
#pragma unroll
  for (int i = 0; i < 2; ++i)
#pragma unroll
    for (int j = 0; j < 4; ++j) acc[i][j] = v8f_zero();

  auto stage_a = [&](int buf, int k0c) {
#ifdef USE_TDM
    if (wave == 0)
      tdm_load_2d(A + (size_t)bm * K + k0c, lds_addr_of(&As[buf][0]), M - bm,
                  K - k0c, K, 64, 128);
#else
#pragma unroll
    for (int i = 0; i < 4; ++i) {
      const int g = tid + i * 256;           // 1024 groups of 8
      const int row = g >> 3, cb = (g & 7) << 3;
      bf16x8 v;
#pragma unroll
      for (int e = 0; e < 8; ++e) v[e] = (__bf16)0.f;
      if (bm + row < M)
        v = *(const bf16x8*)(A + (size_t)(bm + row) * K + k0c + cb);
      *(bf16x8*)(&As[buf][row * 64 + cb]) = v;
    }
#endif
  };
  auto stage_w = [&](int buf, int k0c) {
#pragma unroll
    for (int i = 0; i < 4; ++i) {
      const int g = tid + i * 256;           // 1024 groups of 8
      const int k = g >> 4, nb = (g & 15) << 3;
      bf16x8 v = *(const bf16x8*)(W + (size_t)(k0c + k) * N + bn + nb);
#pragma unroll
      for (int e = 0; e < 8; ++e) Bs[buf][(nb + e) * 72 + k] = v[e];
    }
  };

  stage_a(0, 0);
  stage_w(0, 0);
  int p = 0;
  for (int k0 = 0; k0 < K; k0 += 64) {
    const bool more = (k0 + 64) < K;
    if (more) { stage_a(p ^ 1, k0 + 64); stage_w(p ^ 1, k0 + 64); }
#ifdef USE_TDM
    if (wave == 0) {
      if (more) __builtin_amdgcn_s_wait_tensorcnt(1);
      else      __builtin_amdgcn_s_wait_tensorcnt(0);
    }
#endif
    __syncthreads();
#pragma unroll
    for (int ks = 0; ks < 2; ++ks) {
      bf16x16 a0 = frag_a_lds(&As[p][(wm)*64 + ks * 32], 64);
      bf16x16 a1 = frag_a_lds(&As[p][(wm + 16) * 64 + ks * 32], 64);
#pragma unroll
      for (int j = 0; j < 4; ++j) {
        bf16x16 bb = frag_bt_lds(&Bs[p][(wn + 16 * j) * 72 + ks * 32], 72);
        acc[0][j] = wmma_bf16(a0, bb, acc[0][j]);
        acc[1][j] = wmma_bf16(a1, bb, acc[1][j]);
      }
    }
    __syncthreads();
    p ^= 1;
  }

  const int cn = lane & 15;
  const int rb = (lane >> 4) * 8;
#pragma unroll
  for (int i = 0; i < 2; ++i) {
#pragma unroll
    for (int j = 0; j < 4; ++j) {
      const int col = bn + wn + 16 * j + cn;
#pragma unroll
      for (int r = 0; r < 8; ++r) {
        const int row = bm + wm + 16 * i + rb + r;
        if (row < M) {
          float v = acc[i][j][r];
          if (bias) v += bias[col];
          if (RESID) v += resid[(size_t)row * N + col];
          if (GELU_ACT) v = gelu_exact(v);
          C[(size_t)row * N + col] = (CT)v;
        }
      }
    }
  }
}

// ---------------------------------------------------------------------------
// Flash attention, one (b, head, 128-query tile); qkv in bf16 [B*T,1536],
// head columns h*192 + {q:0..63, k:64..127, v:128..191}.  Q/K tiles via TDM,
// V staged transposed (vectorized 16B global reads).  Output bf16.
// ---------------------------------------------------------------------------
__global__ __launch_bounds__(256) void attn_wmma(
    const __bf16* __restrict__ qkv, __bf16* __restrict__ out) {
  const int qb   = blockIdx.x * 128;
  const int head = blockIdx.y;
  const int b    = blockIdx.z;
  const int tid  = threadIdx.x;
  const int lane = tid & 31;
  const int wave = tid >> 5;

  __shared__ __bf16 Qs[128 * 64];
  __shared__ __bf16 Ks[32 * 64];
  __shared__ __bf16 Vt[64 * 40];     // [c][s], padded
  __shared__ float  Ss[128 * 32];
  __shared__ __bf16 Ps[128 * 32];
  __shared__ float  alpha_s[128];
  __shared__ float  linv_s[128];

  const size_t bh = (size_t)b * 1024;
  const __bf16* qp = qkv + (bh + qb) * 1536 + head * 192;
#ifdef USE_TDM
  if (wave == 0)
    tdm_load_2d(qp, lds_addr_of(&Qs[0]), 128, 64, 1536, 64, 128);
#else
#pragma unroll
  for (int i = 0; i < 4; ++i) {
    const int g = tid + i * 256;           // 1024 groups of 8
    const int row = g >> 3, cb = (g & 7) << 3;
    *(bf16x8*)(&Qs[row * 64 + cb]) =
        *(const bf16x8*)(qp + (size_t)row * 1536 + cb);
  }
#endif

  v8f o[4];
#pragma unroll
  for (int j = 0; j < 4; ++j) o[j] = v8f_zero();
  float mrow = -1e30f, lrow = 0.f;

  for (int s0 = 0; s0 < 1024; s0 += 32) {
    __syncthreads();
    const __bf16* kp = qkv + (bh + s0) * 1536 + head * 192 + 64;
    const __bf16* vp = qkv + (bh + s0) * 1536 + head * 192 + 128;
#ifdef USE_TDM
    if (wave == 0)
      tdm_load_2d(kp, lds_addr_of(&Ks[0]), 32, 64, 1536, 64, 32);
#endif
    {  // V chunk (32x64): one coalesced 16B read per thread, transposed store.
      const int s  = tid >> 3;             // 0..31
      const int cb = (tid & 7) << 3;       // 0..56
#ifndef USE_TDM
      *(bf16x8*)(&Ks[s * 64 + cb]) =
          *(const bf16x8*)(kp + (size_t)s * 1536 + cb);
#endif
      bf16x8 v = *(const bf16x8*)(vp + (size_t)s * 1536 + cb);
#pragma unroll
      for (int e = 0; e < 8; ++e) Vt[(cb + e) * 40 + s] = v[e];
    }
#ifdef USE_TDM
    if (wave == 0) __builtin_amdgcn_s_wait_tensorcnt(0);
#endif
    __syncthreads();

    // S = Q * K^T for this wave's 16 rows x 32 kv-cols.
    v8f sacc[2] = {v8f_zero(), v8f_zero()};
#pragma unroll
    for (int kk = 0; kk < 2; ++kk) {
      bf16x16 a = frag_a_lds(&Qs[wave * 16 * 64 + kk * 32], 64);
#pragma unroll
      for (int j = 0; j < 2; ++j) {
        bf16x16 bt = frag_bt_lds(&Ks[(j * 16) * 64 + kk * 32], 64);
        sacc[j] = wmma_bf16(a, bt, sacc[j]);
      }
    }
#pragma unroll
    for (int j = 0; j < 2; ++j)
#pragma unroll
      for (int r = 0; r < 8; ++r)
        Ss[(wave * 16 + (lane >> 4) * 8 + r) * 32 + j * 16 + (lane & 15)] =
            sacc[j][r];
    __syncthreads();

    // Online softmax: thread t (<128) owns query row t (wave-uniform branch).
    if (tid < 128) {
      float sv[32], mx = -1e30f;
#pragma unroll
      for (int j = 0; j < 32; ++j) {
        sv[j] = Ss[tid * 32 + j] * 0.125f;  // SCALE^2 = 1/sqrt(64)
        mx = fmaxf(mx, sv[j]);
      }
      const float newm  = fmaxf(mrow, mx);
      const float alpha = __expf(mrow - newm);
      float ls = 0.f;
#pragma unroll
      for (int j = 0; j < 32; ++j) {
        const float pv = __expf(sv[j] - newm);
        ls += pv;
        Ps[tid * 32 + j] = (__bf16)pv;
      }
      lrow = lrow * alpha + ls;
      mrow = newm;
      alpha_s[tid] = alpha;
    }
    __syncthreads();

    float al[8];
#pragma unroll
    for (int r = 0; r < 8; ++r)
      al[r] = alpha_s[wave * 16 + (lane >> 4) * 8 + r];
#pragma unroll
    for (int j = 0; j < 4; ++j)
#pragma unroll
      for (int r = 0; r < 8; ++r) o[j][r] *= al[r];
    bf16x16 pa = frag_a_lds(&Ps[wave * 16 * 32], 32);
#pragma unroll
    for (int j = 0; j < 4; ++j) {
      bf16x16 vb = frag_bt_lds(&Vt[(j * 16) * 40], 40);
      o[j] = wmma_bf16(pa, vb, o[j]);
    }
  }

  __syncthreads();
  if (tid < 128) linv_s[tid] = 1.f / lrow;
  __syncthreads();
  float li[8];
#pragma unroll
  for (int r = 0; r < 8; ++r)
    li[r] = linv_s[wave * 16 + (lane >> 4) * 8 + r];
#pragma unroll
  for (int j = 0; j < 4; ++j)
#pragma unroll
    for (int r = 0; r < 8; ++r) {
      const int t = qb + wave * 16 + (lane >> 4) * 8 + r;
      const int c = j * 16 + (lane & 15);
      out[(bh + t) * 512 + head * 64 + c] = (__bf16)(o[j][r] * li[r]);
    }
}

// ---------------------------------------------------------------------------
// LayerNorm over 512-wide rows: one wave32 per row; templated output type so
// GEMM-feeding normalizations store bf16 directly.
// ---------------------------------------------------------------------------
template <typename OT>
__global__ __launch_bounds__(256) void layernorm_512(
    const float* __restrict__ x, const float* __restrict__ g,
    const float* __restrict__ bta, OT* __restrict__ y, int M) {
  const int wave = threadIdx.x >> 5;
  const int lane = threadIdx.x & 31;
  const size_t row = (size_t)blockIdx.x * 8 + wave;
  if (row >= (size_t)M) return;
  const float* xr = x + row * 512 + lane * 16;
  float v[16], s = 0.f;
#pragma unroll
  for (int i = 0; i < 16; ++i) { v[i] = xr[i]; s += v[i]; }
#pragma unroll
  for (int off = 16; off; off >>= 1) s += __shfl_xor(s, off, 32);
  const float mean = s * (1.f / 512.f);
  float vs = 0.f;
#pragma unroll
  for (int i = 0; i < 16; ++i) { const float d = v[i] - mean; vs += d * d; }
#pragma unroll
  for (int off = 16; off; off >>= 1) vs += __shfl_xor(vs, off, 32);
  const float rstd = rsqrtf(vs * (1.f / 512.f) + 1e-5f);
  OT* yr = y + row * 512 + lane * 16;
#pragma unroll
  for (int i = 0; i < 16; ++i) {
    const int c = lane * 16 + i;
    yr[i] = (OT)((v[i] - mean) * rstd * g[c] + bta[c]);
  }
}

__global__ __launch_bounds__(256) void f32_to_bf16(
    const float* __restrict__ s, __bf16* __restrict__ d, size_t n) {
  size_t i = ((size_t)blockIdx.x * 256 + threadIdx.x) * 4;
  if (i + 3 < n) {
    float4 v = *(const float4*)(s + i);
    d[i] = (__bf16)v.x; d[i + 1] = (__bf16)v.y;
    d[i + 2] = (__bf16)v.z; d[i + 3] = (__bf16)v.w;
  }
}

// h[b,t,:] = ce_proj[t,:] + b_in + sum_{c<9} xin[c]*w_in[c,:]
__global__ __launch_bounds__(256) void input_proj(
    const float* __restrict__ x, const float* __restrict__ params,
    const float* __restrict__ ceproj, const float* __restrict__ w_in,
    const float* __restrict__ b_in, float* __restrict__ h) {
  const int t = blockIdx.x, b = blockIdx.y;
  __shared__ float sc[9];
  if (threadIdx.x < 3)
    sc[threadIdx.x] = x[((size_t)b * 3 + threadIdx.x) * 1024 + t];
  else if (threadIdx.x < 9)
    sc[threadIdx.x] = params[((size_t)b * 6 + (threadIdx.x - 3)) * 1024 + t];
  __syncthreads();
#pragma unroll
  for (int i = 0; i < 2; ++i) {
    const int w = threadIdx.x + i * 256;
    float acc = ceproj[(size_t)t * 512 + w] + b_in[w];
#pragma unroll
    for (int c = 0; c < 9; ++c) acc += sc[c] * w_in[c * 512 + w];
    h[((size_t)b * 1024 + t) * 512 + w] = acc;
  }
}

// outh[b,w,t] = hout[b,t,w] * mask[b,t]
__global__ __launch_bounds__(256) void transpose_mask(
    const float* __restrict__ hout, const float* __restrict__ mask,
    float* __restrict__ outh) {
  __shared__ float tile[64][65];
  const int ws = blockIdx.x * 64, ts = blockIdx.y * 64, b = blockIdx.z;
#pragma unroll
  for (int i = 0; i < 16; ++i) {
    const int idx = threadIdx.x + i * 256;
    const int r = idx >> 6, c = idx & 63;
    tile[r][c] = hout[((size_t)b * 1024 + ts + r) * 512 + ws + c] *
                 mask[(size_t)b * 1024 + ts + r];
  }
  __syncthreads();
#pragma unroll
  for (int i = 0; i < 16; ++i) {
    const int idx = threadIdx.x + i * 256;
    const int r = idx >> 6, c = idx & 63;
    outh[((size_t)b * 512 + ws + r) * 1024 + ts + c] = tile[c][r];
  }
}

// pooled[b,w] = max_t( hout[b,t,w]*mask + (1-mask)*NEG )  -> bf16 for agg GEMM
__global__ __launch_bounds__(256) void pool_max(
    const float* __restrict__ hout, const float* __restrict__ mask,
    __bf16* __restrict__ pooled) {
  const int b = blockIdx.y;
  const int w = blockIdx.x * 256 + threadIdx.x;
  float mx = -3.4e38f;
  for (int t = 0; t < 1024; ++t) {
    const float mk = mask[(size_t)b * 1024 + t];
    const float v = hout[((size_t)b * 1024 + t) * 512 + w] * mk +
                    (1.f - mk) * (-100000.f);
    mx = fmaxf(mx, v);
  }
  pooled[(size_t)b * 512 + w] = (__bf16)mx;
}

// ---------------------------------------------------------------------------
// Host orchestration.
// ---------------------------------------------------------------------------
static inline void conv(const float* s, __bf16* d, size_t n,
                        hipStream_t stream) {
  f32_to_bf16<<<dim3((unsigned)((n / 4 + 255) / 256)), 256, 0, stream>>>(s, d,
                                                                         n);
}

extern "C" void kernel_launch(void* const* d_in, const int* in_sizes, int n_in,
                              void* d_out, int out_size, void* d_ws,
                              size_t ws_size, hipStream_t stream) {
  (void)in_sizes; (void)n_in; (void)out_size; (void)ws_size;
  const float* x       = (const float*)d_in[0];
  const float* params  = (const float*)d_in[1];
  const float* mask    = (const float*)d_in[2];
  const float* cemb    = (const float*)d_in[3];
  const float* w_in    = (const float*)d_in[4];
  const float* b_in    = (const float*)d_in[5];
  const float* lpre_g  = (const float*)d_in[6];
  const float* lpre_b  = (const float*)d_in[7];
  const float* qkv_w   = (const float*)d_in[8];
  const float* qkv_b   = (const float*)d_in[9];
  const float* pw      = (const float*)d_in[10];
  const float* pb      = (const float*)d_in[11];
  const float* l1g     = (const float*)d_in[12];
  const float* l1b     = (const float*)d_in[13];
  const float* fcw     = (const float*)d_in[14];
  const float* fcb     = (const float*)d_in[15];
  const float* fpw     = (const float*)d_in[16];
  const float* fpb     = (const float*)d_in[17];
  const float* l2g     = (const float*)d_in[18];
  const float* l2b     = (const float*)d_in[19];
  const float* lpost_g = (const float*)d_in[20];
  const float* lpost_b = (const float*)d_in[21];
  const float* w_out   = (const float*)d_in[22];
  const float* b_out   = (const float*)d_in[23];
  const float* afw     = (const float*)d_in[24];
  const float* afb     = (const float*)d_in[25];
  const float* apw     = (const float*)d_in[26];
  const float* apb     = (const float*)d_in[27];

  const size_t CE  = (size_t)1024 * 512;
  const size_t HT  = (size_t)32 * 1024 * 512;
  const size_t BIG = (size_t)32 * 1024 * 2048;

  // fp32 region
  float* ce   = (float*)d_ws;
  float* h    = ce + CE;
  float* hout = h + HT;           // also input_proj temp
  // bf16 region
  __bf16* hn_bf   = (__bf16*)(hout + HT);
  __bf16* big_bf  = hn_bf + HT;                 // qkv / ff (disjoint in time)
  __bf16* ao_bf   = big_bf + BIG;               // attention output
  __bf16* pooled  = ao_bf + HT;
  __bf16* amid    = pooled + (size_t)128 * 512;
  __bf16* cemb_bf = amid + (size_t)128 * 2048;
  __bf16* win_bf  = cemb_bf + (size_t)1024 * 1024;
  __bf16* qkvw_bf = win_bf + (size_t)1024 * 512;
  __bf16* pw_bf   = qkvw_bf + (size_t)12 * 512 * 1536;
  __bf16* fcw_bf  = pw_bf + (size_t)12 * 512 * 512;
  __bf16* fpw_bf  = fcw_bf + (size_t)12 * 512 * 2048;
  __bf16* wout_bf = fpw_bf + (size_t)12 * 2048 * 512;
  __bf16* afw_bf  = wout_bf + (size_t)512 * 512;
  __bf16* apw_bf  = afw_bf + (size_t)512 * 2048;

  float* zout  = (float*)d_out;
  float* h_out = zout + (size_t)32 * 512;

  const int MT = 32 * 1024;

  // 0) weights -> bf16 arena (cheap: ~83 MB @ 23.3 TB/s).
  conv(cemb, cemb_bf, (size_t)1024 * 1024, stream);
  conv(w_in + 9 * 512, win_bf, (size_t)1024 * 512, stream);
  conv(qkv_w, qkvw_bf, (size_t)12 * 512 * 1536, stream);
  conv(pw, pw_bf, (size_t)12 * 512 * 512, stream);
  conv(fcw, fcw_bf, (size_t)12 * 512 * 2048, stream);
  conv(fpw, fpw_bf, (size_t)12 * 2048 * 512, stream);
  conv(w_out, wout_bf, (size_t)512 * 512, stream);
  conv(afw, afw_bf, (size_t)512 * 2048, stream);
  conv(apw, apw_bf, (size_t)512 * 2048, stream);

  // 1) batch-independent class-embedding projection.
  gemm_wmma<false, false, float>
      <<<dim3(4, 8), 256, 0, stream>>>(cemb_bf, win_bf, nullptr, nullptr, ce,
                                       1024, 1024, 512);

  // 2) input projection (K=9 part) + ln_pre.
  input_proj<<<dim3(1024, 32), 256, 0, stream>>>(x, params, ce, w_in, b_in,
                                                 hout);
  layernorm_512<float><<<MT / 8, 256, 0, stream>>>(hout, lpre_g, lpre_b, h,
                                                   MT);

  // 3) transformer blocks.
  for (int l = 0; l < 12; ++l) {
    layernorm_512<__bf16><<<MT / 8, 256, 0, stream>>>(
        h, l1g + l * 512, l1b + l * 512, hn_bf, MT);
    gemm_wmma<false, false, __bf16><<<dim3(12, 256), 256, 0, stream>>>(
        hn_bf, qkvw_bf + (size_t)l * 512 * 1536, qkv_b + (size_t)l * 1536,
        nullptr, big_bf, MT, 512, 1536);
    attn_wmma<<<dim3(8, 8, 32), 256, 0, stream>>>(big_bf, ao_bf);
    gemm_wmma<false, true, float><<<dim3(4, 256), 256, 0, stream>>>(
        ao_bf, pw_bf + (size_t)l * 512 * 512, pb + (size_t)l * 512, h, h, MT,
        512, 512);
    layernorm_512<__bf16><<<MT / 8, 256, 0, stream>>>(
        h, l2g + l * 512, l2b + l * 512, hn_bf, MT);
    gemm_wmma<true, false, __bf16><<<dim3(16, 256), 256, 0, stream>>>(
        hn_bf, fcw_bf + (size_t)l * 512 * 2048, fcb + (size_t)l * 2048,
        nullptr, big_bf, MT, 512, 2048);
    gemm_wmma<false, true, float><<<dim3(4, 256), 256, 0, stream>>>(
        big_bf, fpw_bf + (size_t)l * 2048 * 512, fpb + (size_t)l * 512, h, h,
        MT, 2048, 512);
  }

  // 4) ln_post + output projection.
  layernorm_512<__bf16><<<MT / 8, 256, 0, stream>>>(h, lpost_g, lpost_b, hn_bf,
                                                    MT);
  gemm_wmma<false, false, float><<<dim3(4, 256), 256, 0, stream>>>(
      hn_bf, wout_bf, b_out, nullptr, hout, MT, 512, 512);

  // 5) masked transpose (second output) + masked max pool.
  transpose_mask<<<dim3(8, 16, 32), 256, 0, stream>>>(hout, mask, h_out);
  pool_max<<<dim3(2, 32), 256, 0, stream>>>(hout, mask, pooled);

  // 6) aggregation MLP (M=32; TDM zero-fills rows >= M, stores are masked).
  gemm_wmma<true, false, __bf16><<<dim3(16, 1), 256, 0, stream>>>(
      pooled, afw_bf, afb, nullptr, amid, 32, 512, 2048);
  gemm_wmma<false, false, float><<<dim3(4, 1), 256, 0, stream>>>(
      amid, apw_bf, apb, nullptr, zout, 32, 2048, 512);
}
